// GIG_Model_12524124635752
// MI455X (gfx1250) — matrix operand order repaired and checked by reading, not compiled, not run on hardware
//
#include <hip/hip_runtime.h>
#include <hip/hip_bf16.h>
#include <stdint.h>

typedef _Float16 half_t;
typedef __attribute__((ext_vector_type(16))) _Float16     v16h;
typedef __attribute__((ext_vector_type(8)))  float        v8f;
typedef __attribute__((ext_vector_type(8)))  unsigned int v8u;

#define NIMG 4096
#define PIXN 49
#define NPIX (NIMG * PIXN)          /* 200704 */
#define PIX_LAST (NPIX - 1)         /* 200703 */
#define PIX_SYM  ((NIMG - 1) * PIXN)/* 200655 */

// ---------------------------------------------------------------------------
// Weight prep: conv weights [O][K] f32 -> paired-K f16: dst[(k/2)*O + n] packs
// halves (k, k+1) of column n into one u32 (coalesced B-fragment loads).
// ---------------------------------------------------------------------------
__global__ __launch_bounds__(256) void k_prep_wpair(const float* __restrict__ src,
                                                    unsigned int* __restrict__ dst,
                                                    int K, int O) {
  int i = blockIdx.x * blockDim.x + threadIdx.x;
  int total = (K >> 1) * O;
  if (i >= total) return;
  int ku = i / O, n = i - ku * O;
  int k0 = ku * 2;
  half_t lo = (half_t)src[(size_t)n * K + k0];
  half_t hi = (half_t)src[(size_t)n * K + k0 + 1];
  unsigned int l = __builtin_bit_cast(unsigned short, lo);
  unsigned int h = __builtin_bit_cast(unsigned short, hi);
  dst[i] = l | (h << 16);
}

__global__ __launch_bounds__(256) void k_cvt_h(const float* __restrict__ src,
                                               half_t* __restrict__ dst, int n) {
  int i = blockIdx.x * blockDim.x + threadIdx.x;
  if (i < n) dst[i] = (half_t)src[i];
}

// ---------------------------------------------------------------------------
// 3x3 SAME conv on 7x7 images as implicit GEMM with WMMA.
// One workgroup (256 thr = 8 waves) per image. Rows = 49 pixels (padded to 64),
// cols = COUT, K = CIN*9.  rt = wave&3, col group = wave>>2 (TPW tiles each).
// ---------------------------------------------------------------------------
template <bool INF32, int CIN, int COUT, int TPW, bool RELU, bool NODEOUT>
__global__ __launch_bounds__(256) void k_conv3x3(const void* __restrict__ inp,
                                                 const unsigned int* __restrict__ wtp,
                                                 const float* __restrict__ bias,
                                                 half_t* __restrict__ outp) {
  __shared__ half_t img[CIN * 81];  // zero-padded 9x9 per channel
  const int t = threadIdx.x;
  const int im = blockIdx.x;

  for (int i = t; i < CIN * 81; i += 256) img[i] = (half_t)0.0f;
  __syncthreads();
  for (int i = t; i < CIN * 49; i += 256) {
    int cin = i / 49, p = i - cin * 49;
    int y = p / 7, x = p - y * 7;
    float v;
    if (INF32) v = ((const float*)inp)[(size_t)im * CIN * 49 + i];
    else       v = (float)((const half_t*)inp)[(size_t)im * CIN * 49 + i];
    img[cin * 81 + (y + 1) * 9 + (x + 1)] = (half_t)v;
  }
  __syncthreads();

  const int wid = t >> 5, lane = t & 31;
  const int rt = wid & 3;
  const int ctBase = (wid >> 2) * TPW;
  const int half_ = lane >> 4, l15 = lane & 15;

  v8f acc[TPW];
#pragma unroll
  for (int j = 0; j < TPW; ++j)
#pragma unroll
    for (int r = 0; r < 8; ++r) acc[j][r] = 0.0f;

  const int pix = rt * 16 + l15;      // A-matrix row (output pixel)
  const int pv = pix < 49 ? pix : 0;
  const int py = pv / 7, px = pv - py * 7;

  const int K = CIN * 9;
  for (int kc = 0; kc < K; kc += 32) {
    // prefetch next weight chunk (L2-hot reuse stream)
    if (kc + 32 < K) {
      int kg = kc + 32 + half_ * 16;
      __builtin_prefetch(wtp + (size_t)(kg >> 1) * COUT + (ctBase * 16 + l15), 0, 0);
    }
    // A fragment: 16-bit A 16x32 layout (ISA 7.12.2)
    v16h a;
#pragma unroll
    for (int v = 0; v < 8; ++v) {
      int kb = kc + ((v & 3) * 2) + ((v >> 2) * 16) + half_ * 8;
#pragma unroll
      for (int s = 0; s < 2; ++s) {
        int kg = kb + s;
        int cin = kg / 9;
        int r = kg - cin * 9;
        int ky = r / 3, kx = r - ky * 3;
        half_t val = img[cin * 81 + (py + ky) * 9 + (px + kx)];
        a[v * 2 + s] = (pix < 49) ? val : (half_t)0.0f;
      }
    }
#pragma unroll
    for (int j = 0; j < TPW; ++j) {
      int n = (ctBase + j) * 16 + l15;  // B column
      v8u bu;
#pragma unroll
      for (int v = 0; v < 8; ++v) {
        int kg = kc + ((v & 3) * 2) + ((v >> 2) * 16) + half_ * 8;  // even
        bu[v] = wtp[(size_t)(kg >> 1) * COUT + n];     // halves (kg, kg+1)
      }
      v16h b = __builtin_bit_cast(v16h, bu);
      acc[j] = __builtin_amdgcn_wmma_f32_16x16x32_f16(false, a, false, b,
                                                      (short)0, acc[j], false, false);
    }
  }

#pragma unroll
  for (int j = 0; j < TPW; ++j) {
    int n = (ctBase + j) * 16 + l15;
    float bv = bias[n];
#pragma unroll
    for (int r = 0; r < 8; ++r) {
      int p = rt * 16 + r + 8 * half_;
      if (p < 49) {
        float v = acc[j][r] + bv;
        if (RELU) v = fmaxf(v, 0.0f);
        if (NODEOUT) outp[((size_t)im * 49 + p) * COUT + n] = (half_t)v;   // [node][C]
        else         outp[((size_t)im * COUT + n) * 49 + p] = (half_t)v;   // [img][C][49]
      }
    }
  }
}

// ---------------------------------------------------------------------------
// Generic f16 GEMM: C[M,N] f32 = A[M,K] f16 * B[K,N] f16.
// M % 64 == 0, N % 64 == 0, K % 32 == 0.  256 threads, 64x64 tile per block.
// A tile staged global->LDS with GLOBAL_LOAD_ASYNC_TO_LDS_B32 (ASYNCcnt);
// B tile staged transposed through VGPRs.  Each wave: 1 A frag, 2 B frags,
// 2 wmma per 32-wide K chunk.
// ---------------------------------------------------------------------------
__global__ __launch_bounds__(256) void k_gemm(const half_t* __restrict__ A,
                                              const half_t* __restrict__ B,
                                              float* __restrict__ C,
                                              int M, int N, int K) {
  __shared__ half_t As[64 * 40];   // [row][k] stride 40
  __shared__ half_t Bs[64 * 40];   // transposed: [n][k] stride 40
  const int t = threadIdx.x;
  const int mBase = blockIdx.x * 64;
  const int nBase = blockIdx.y * 64;
  const int wid = t >> 5, lane = t & 31;
  const int rt = wid & 3, cg = wid >> 2;          // 4 row tiles x 2 col groups
  const int half_ = lane >> 4, l15 = lane & 15;
  (void)M;

  v8f acc[2];
#pragma unroll
  for (int j = 0; j < 2; ++j)
#pragma unroll
    for (int r = 0; r < 8; ++r) acc[j][r] = 0.0f;

  // Per-thread async A-staging addresses (loop-invariant; only SADDR advances).
  // LDS aperture maps addr[31:0] directly to the LDS offset (ISA 10.2).
  const int arow = t >> 4;          // base row 0..15 (+16 per i)
  const int acu  = t & 15;          // u32 column within 32-half row
  unsigned ldsA[4], voffA[4];
#pragma unroll
  for (int i = 0; i < 4; ++i) {
    int row = arow + i * 16;
    ldsA[i]  = (unsigned)(size_t)(As + row * 40 + acu * 2);
    voffA[i] = (unsigned)((row * K + acu * 2) * 2);   // byte offset from abase
  }
  const half_t* abase = A + (size_t)mBase * K;

  for (int k0 = 0; k0 < K; k0 += 32) {
    // ---- async copy A tile: 1024 dwords, 4 per lane, no VGPR round-trip
#pragma unroll
    for (int i = 0; i < 4; ++i) {
      asm volatile("global_load_async_to_lds_b32 %0, %1, %2"
                   :: "v"(ldsA[i]), "v"(voffA[i]), "s"(abase)
                   : "memory");
    }
    // ---- stage B tile transposed: 32 k-rows x 64 cols
#pragma unroll
    for (int i = 0; i < 4; ++i) {
      int u = t + i * 256;
      int row = u >> 5, cu = u & 31;
      unsigned int val = *(const unsigned int*)(B + (size_t)(k0 + row) * N + nBase + cu * 2);
      Bs[(cu * 2 + 0) * 40 + row] = __builtin_bit_cast(half_t, (unsigned short)(val & 0xffffu));
      Bs[(cu * 2 + 1) * 40 + row] = __builtin_bit_cast(half_t, (unsigned short)(val >> 16));
    }
    if (k0 + 32 < K)  // prefetch next B chunk
      __builtin_prefetch(B + (size_t)(k0 + 32 + (t >> 5)) * N + nBase + (t & 31) * 2, 0, 0);

    asm volatile("s_wait_asynccnt 0" ::: "memory");
    __syncthreads();

    v8u au;
#pragma unroll
    for (int v = 0; v < 8; ++v) {
      int ka = ((v & 3) * 2) + ((v >> 2) * 16) + half_ * 8;          // A layout
      au[v] = *(const unsigned int*)(As + (rt * 16 + l15) * 40 + ka);
    }
    v16h a = __builtin_bit_cast(v16h, au);
#pragma unroll
    for (int j = 0; j < 2; ++j) {
      int ct = cg * 2 + j;
      v8u bu;
#pragma unroll
      for (int v = 0; v < 8; ++v) {
        int kb = ((v & 3) * 2) + ((v >> 2) * 16) + half_ * 8;        // B layout
        bu[v] = *(const unsigned int*)(Bs + (ct * 16 + l15) * 40 + kb);
      }
      v16h b = __builtin_bit_cast(v16h, bu);
      acc[j] = __builtin_amdgcn_wmma_f32_16x16x32_f16(false, a, false, b,
                                                      (short)0, acc[j], false, false);
    }
    __syncthreads();   // all frag reads done before next chunk's async writes
    abase += 32;
  }

#pragma unroll
  for (int j = 0; j < 2; ++j) {   // D layout: M=r+8*half, N=lane&15
    int ct = cg * 2 + j;
#pragma unroll
    for (int r = 0; r < 8; ++r) {
      int m = mBase + rt * 16 + r + 8 * half_;
      int n = nBase + ct * 16 + l15;
      C[(size_t)m * N + n] = acc[j][r];
    }
  }
}

// ---------------------------------------------------------------------------
// Pixel-graph GCN aggregation (grid 7x7 per image, edges duplicated, one self
// loop, trailing last<->sym edge, plus PyG extra self loop xw/deg, bias, relu).
// ---------------------------------------------------------------------------
__device__ __forceinline__ float pix_dis(int node) {
  int p = node % 49;
  int x = p % 7, y = p / 7;
  int nnb = (x > 0) + (x < 6) + (y > 0) + (y < 6);
  int deg = 2 + 2 * nnb + ((node == PIX_LAST) || (node == PIX_SYM) ? 1 : 0);
  return rsqrtf((float)deg);
}

template <bool RELU>
__global__ __launch_bounds__(128) void k_pixagg(const float* __restrict__ xw,
                                                const float* __restrict__ bias,
                                                half_t* __restrict__ out, int C) {
  int node = blockIdx.x;
  int p = node % 49;
  int x = p % 7, y = p / 7;
  int nnb = (x > 0) + (x < 6) + (y > 0) + (y < 6);
  int deg = 2 + 2 * nnb + ((node == PIX_LAST) || (node == PIX_SYM) ? 1 : 0);
  float di = rsqrtf((float)deg);
  float wself = di * di + 1.0f / (float)deg;
  float wl = (x > 0) ? 2.0f * di * pix_dis(node - 1) : 0.0f;
  float wr = (x < 6) ? 2.0f * di * pix_dis(node + 1) : 0.0f;
  float wu = (y > 0) ? 2.0f * di * pix_dis(node - 7) : 0.0f;
  float wd = (y < 6) ? 2.0f * di * pix_dis(node + 7) : 0.0f;
  float wsp = 0.0f; int spn = 0;
  if (node == PIX_LAST) { wsp = di * pix_dis(PIX_SYM);  spn = PIX_SYM;  }
  if (node == PIX_SYM)  { wsp = di * pix_dis(PIX_LAST); spn = PIX_LAST; }

  for (int c = threadIdx.x; c < C; c += blockDim.x) {
    float s = wself * xw[(size_t)node * C + c];
    if (x > 0) s += wl * xw[(size_t)(node - 1) * C + c];
    if (x < 6) s += wr * xw[(size_t)(node + 1) * C + c];
    if (y > 0) s += wu * xw[(size_t)(node - 7) * C + c];
    if (y < 6) s += wd * xw[(size_t)(node + 7) * C + c];
    if (wsp != 0.0f) s += wsp * xw[(size_t)spn * C + c];
    s += bias[c];
    if (RELU) s = fmaxf(s, 0.0f);
    out[(size_t)node * C + c] = (half_t)s;
  }
}

// ---------------------------------------------------------------------------
// Chain-graph GCN aggregation (4096 nodes, duplicated neighbor edges).
// ---------------------------------------------------------------------------
__device__ __forceinline__ float chain_dis(int i, int Nn) {
  int nnb = (i > 0) + (i < Nn - 1);
  return rsqrtf((float)(2 + 2 * nnb));
}

template <bool RELU, typename OT>
__global__ __launch_bounds__(256) void k_chainagg(const float* __restrict__ xw,
                                                  const float* __restrict__ bias,
                                                  OT* __restrict__ out, int C, int Nn) {
  int node = blockIdx.x;
  int nnb = (node > 0) + (node < Nn - 1);
  float deg = (float)(2 + 2 * nnb);
  float di = rsqrtf(deg);
  float wself = di * di + 1.0f / deg;
  float wl = (node > 0)      ? 2.0f * di * chain_dis(node - 1, Nn) : 0.0f;
  float wr = (node < Nn - 1) ? 2.0f * di * chain_dis(node + 1, Nn) : 0.0f;
  for (int c = threadIdx.x; c < C; c += blockDim.x) {
    float s = wself * xw[(size_t)node * C + c];
    if (node > 0)      s += wl * xw[(size_t)(node - 1) * C + c];
    if (node < Nn - 1) s += wr * xw[(size_t)(node + 1) * C + c];
    s += bias[c];
    if (RELU) s = fmaxf(s, 0.0f);
    out[(size_t)node * C + c] = (OT)s;
  }
}

// ---------------------------------------------------------------------------
// Classifier: logits[row] = bout[row] @ wc + bc   (wc: [3136][2] f32)
// ---------------------------------------------------------------------------
__global__ __launch_bounds__(256) void k_classifier(const float* __restrict__ xb,
                                                    const float* __restrict__ wc,
                                                    const float* __restrict__ bc,
                                                    float* __restrict__ out) {
  __shared__ float red0[256], red1[256];
  int row = blockIdx.x, t = threadIdx.x;
  float s0 = 0.0f, s1 = 0.0f;
  for (int k = t; k < 3136; k += 256) {
    float v = xb[(size_t)row * 3136 + k];
    s0 += v * wc[k * 2 + 0];
    s1 += v * wc[k * 2 + 1];
  }
  red0[t] = s0; red1[t] = s1;
  __syncthreads();
  for (int off = 128; off > 0; off >>= 1) {
    if (t < off) { red0[t] += red0[t + off]; red1[t] += red1[t + off]; }
    __syncthreads();
  }
  if (t == 0) {
    out[row * 2 + 0] = red0[0] + bc[0];
    out[row * 2 + 1] = red1[0] + bc[1];
  }
}

// ---------------------------------------------------------------------------
extern "C" void kernel_launch(void* const* d_in, const int* in_sizes, int n_in,
                              void* d_out, int out_size, void* d_ws, size_t ws_size,
                              hipStream_t stream) {
  (void)in_sizes; (void)n_in; (void)out_size; (void)ws_size;

  const float* features = (const float*)d_in[0];
  const float* cw1 = (const float*)d_in[1];  const float* cb1 = (const float*)d_in[2];
  const float* cw2 = (const float*)d_in[3];  const float* cb2 = (const float*)d_in[4];
  const float* cw3 = (const float*)d_in[5];  const float* cb3 = (const float*)d_in[6];
  const float* wp1 = (const float*)d_in[7];  const float* bp1 = (const float*)d_in[8];
  const float* wp2 = (const float*)d_in[9];  const float* bp2 = (const float*)d_in[10];
  const float* wp3 = (const float*)d_in[11]; const float* bp3 = (const float*)d_in[12];
  const float* wb1 = (const float*)d_in[13]; const float* bb1 = (const float*)d_in[14];
  const float* wb2 = (const float*)d_in[15]; const float* bb2 = (const float*)d_in[16];
  const float* wb3 = (const float*)d_in[17]; const float* bb3 = (const float*)d_in[18];
  const float* wc  = (const float*)d_in[19]; const float* bc  = (const float*)d_in[20];

  char* ws = (char*)d_ws;
  size_t off = 0;
  auto alloc = [&](size_t bytes) -> void* {
    void* p = ws + off;
    off += (bytes + 255) & ~(size_t)255;
    return p;
  };

  unsigned int* wt1p = (unsigned int*)alloc((size_t)1152 * 128 * 4);  // conv1 K=2304
  unsigned int* wt2p = (unsigned int*)alloc((size_t)576 * 64 * 4);    // conv2 K=1152
  unsigned int* wt3p = (unsigned int*)alloc((size_t)288 * 64 * 4);    // conv3 K=576
  half_t* wp1h = (half_t*)alloc((size_t)64 * 128 * 2);
  half_t* wp2h = (half_t*)alloc((size_t)128 * 128 * 2);
  half_t* wp3h = (half_t*)alloc((size_t)128 * 64 * 2);
  half_t* wb1h = (half_t*)alloc((size_t)3136 * 64 * 2);
  half_t* wb2h = (half_t*)alloc((size_t)64 * 64 * 2);
  half_t* wb3h = (half_t*)alloc((size_t)64 * 3136 * 2);
  half_t* bufA = (half_t*)alloc((size_t)NIMG * 128 * 49 * 2);   // act1 / nodef2
  half_t* bufB = (half_t*)alloc((size_t)NIMG * 64 * 49 * 2);    // act2 / batchfeat
  half_t* bufC = (half_t*)alloc((size_t)NPIX * 64 * 2);         // nodef0
  float*  bufD = (float*)alloc((size_t)NPIX * 128 * 4);         // xw scratch (+bout tail)
  half_t* bufE = (half_t*)alloc((size_t)NPIX * 128 * 2);        // nodef1
  half_t* bf1  = (half_t*)alloc((size_t)NIMG * 64 * 2);
  half_t* bf2  = (half_t*)alloc((size_t)NIMG * 64 * 2);
  float*  bout = bufD + (size_t)NIMG * 3136;                    // second half of bufD

  // ---- weight prep
  k_prep_wpair<<<(1152 * 128 + 255) / 256, 256, 0, stream>>>(cw1, wt1p, 2304, 128);
  k_prep_wpair<<<(576 * 64 + 255) / 256, 256, 0, stream>>>(cw2, wt2p, 1152, 64);
  k_prep_wpair<<<(288 * 64 + 255) / 256, 256, 0, stream>>>(cw3, wt3p, 576, 64);
  k_cvt_h<<<(64 * 128 + 255) / 256, 256, 0, stream>>>(wp1, wp1h, 64 * 128);
  k_cvt_h<<<(128 * 128 + 255) / 256, 256, 0, stream>>>(wp2, wp2h, 128 * 128);
  k_cvt_h<<<(128 * 64 + 255) / 256, 256, 0, stream>>>(wp3, wp3h, 128 * 64);
  k_cvt_h<<<(3136 * 64 + 255) / 256, 256, 0, stream>>>(wb1, wb1h, 3136 * 64);
  k_cvt_h<<<(64 * 64 + 255) / 256, 256, 0, stream>>>(wb2, wb2h, 64 * 64);
  k_cvt_h<<<(64 * 3136 + 255) / 256, 256, 0, stream>>>(wb3, wb3h, 64 * 3136);

  // ---- convs (implicit GEMM, WMMA)
  k_conv3x3<true, 256, 128, 4, true, false><<<NIMG, 256, 0, stream>>>(features, wt1p, cb1, bufA);
  k_conv3x3<false, 128, 64, 2, true, false><<<NIMG, 256, 0, stream>>>(bufA, wt2p, cb2, bufB);
  k_conv3x3<false, 64, 64, 2, false, true><<<NIMG, 256, 0, stream>>>(bufB, wt3p, cb3, bufC);

  // ---- pixel GCN (3 layers): gemm -> analytic aggregation
  { dim3 g(NPIX / 64, 128 / 64);
    k_gemm<<<g, 256, 0, stream>>>(bufC, wp1h, bufD, NPIX, 128, 64); }
  k_pixagg<true><<<NPIX, 128, 0, stream>>>(bufD, bp1, bufE, 128);

  { dim3 g(NPIX / 64, 128 / 64);
    k_gemm<<<g, 256, 0, stream>>>(bufE, wp2h, bufD, NPIX, 128, 128); }
  k_pixagg<true><<<NPIX, 128, 0, stream>>>(bufD, bp2, bufA, 128);

  { dim3 g(NPIX / 64, 64 / 64);
    k_gemm<<<g, 256, 0, stream>>>(bufA, wp3h, bufD, NPIX, 64, 128); }
  k_pixagg<false><<<NPIX, 64, 0, stream>>>(bufD, bp3, bufB, 64);   // -> batchfeat [4096][3136]

  // ---- batch GCN (3 layers) on chain graph
  { dim3 g(NIMG / 64, 64 / 64);
    k_gemm<<<g, 256, 0, stream>>>(bufB, wb1h, bufD, NIMG, 64, 3136); }
  k_chainagg<true, half_t><<<NIMG, 64, 0, stream>>>(bufD, bb1, bf1, 64, NIMG);

  { dim3 g(NIMG / 64, 64 / 64);
    k_gemm<<<g, 256, 0, stream>>>(bf1, wb2h, bufD, NIMG, 64, 64); }
  k_chainagg<true, half_t><<<NIMG, 64, 0, stream>>>(bufD, bb2, bf2, 64, NIMG);

  { dim3 g(NIMG / 64, 3136 / 64);
    k_gemm<<<g, 256, 0, stream>>>(bf2, wb3h, bufD, NIMG, 3136, 64); }
  k_chainagg<false, float><<<NIMG, 256, 0, stream>>>(bufD, bb3, bout, 3136, NIMG);

  // ---- classifier
  k_classifier<<<NIMG, 256, 0, stream>>>(bout, wc, bc, (float*)d_out);
}
